// Conv2d_22668837388707
// MI455X (gfx1250) — matrix operand order repaired and checked by reading, not compiled, not run on hardware
//
#include <hip/hip_runtime.h>
#include <hip/hip_bf16.h>

typedef __bf16 bf16_t;
typedef __attribute__((ext_vector_type(16))) __bf16 v16bf;
typedef __attribute__((ext_vector_type(8)))  float  v8f;

union FragBF { uint4 q[2]; v16bf v; };

#define B_   16
#define HW   4096
#define COUT 128

// ---------------------------------------------------------------------------
// Convert conv weights [Cout][Cin*9] f32 -> chunk-major bf16 [K/32][Cout][32]
// ---------------------------------------------------------------------------
__global__ void convert_w(const float* __restrict__ w, bf16_t* __restrict__ o,
                          int Cout, int Kdim) {
  int gid = blockIdx.x * 256 + threadIdx.x;
  int total = Cout * Kdim;
  if (gid >= total) return;
  int t  = gid & 31;
  int mk = gid >> 5;
  int m  = mk % Cout;
  int kc = mk / Cout;
  o[gid] = (bf16_t)w[(size_t)m * Kdim + kc * 32 + t];
}

// ---------------------------------------------------------------------------
// Direct 3x3 conv producing 18 offset channels + 9 sigmoid(mask) channels
// ---------------------------------------------------------------------------
__launch_bounds__(256)
__global__ void offmod_conv(const float* __restrict__ in,
                            const float* __restrict__ woff,
                            const float* __restrict__ wmod,
                            float* __restrict__ off_out,
                            float* __restrict__ msk_out,
                            int Cin) {
  int gid  = blockIdx.x * 256 + threadIdx.x;
  int pix  = gid & (HW - 1);
  int rest = gid >> 12;
  int c    = rest % 27;
  int b    = rest / 27;
  if (b >= B_) return;
  int y = pix >> 6, x = pix & 63;
  const float* wsel = (c < 18) ? (woff + (size_t)c * Cin * 9)
                               : (wmod + (size_t)(c - 18) * Cin * 9);
  float acc = 0.0f;
  for (int cin = 0; cin < Cin; ++cin) {
    const float* img = in + ((size_t)b * Cin + cin) * HW;
    const float* wk  = wsel + cin * 9;
#pragma unroll
    for (int kh = 0; kh < 3; ++kh) {
      int yy = y - 1 + kh;
      if (yy < 0 || yy > 63) continue;
#pragma unroll
      for (int kw = 0; kw < 3; ++kw) {
        int xx = x - 1 + kw;
        if (xx < 0 || xx > 63) continue;
        acc += img[yy * 64 + xx] * wk[kh * 3 + kw];
      }
    }
  }
  if (c < 18) off_out[((size_t)(b * 18 + c)) * HW + pix] = acc;
  else        msk_out[((size_t)(b * 9 + (c - 18))) * HW + pix] =
                  1.0f / (1.0f + expf(-acc));
}

// ---------------------------------------------------------------------------
// Fused deformable-conv GEMM: per block = (batch b, image row y).
//   D[128 x 64px] = W[128 x Cin*9] * Gather[Cin*9 x 64px]   via bf16 WMMA
// Bilinear geometry precomputed per (k,pixel); wave-uniform (cin,k) kept in
// SGPRs via readfirstlane so corner loads use scalar-base addressing.
// Ping-pong B tile, one barrier per K-chunk. Epilogue: BN (+optional relu).
// ---------------------------------------------------------------------------
__launch_bounds__(256)
__global__ void deform_wmma(const float* __restrict__ xin,
                            const float* __restrict__ offp,
                            const float* __restrict__ mskp,
                            const bf16_t* __restrict__ wbf,
                            const float* __restrict__ bng,
                            const float* __restrict__ bnb,
                            const float* __restrict__ bnm,
                            const float* __restrict__ bnv,
                            float* __restrict__ out,
                            int Cin, int nchunk, int relu) {
  const int tid  = threadIdx.x;
  const int lane = tid & 31;
  const int wave = tid >> 5;            // 0..7 : M-strip of 16 rows
  const int b    = blockIdx.x >> 6;
  const int y    = blockIdx.x & 63;
  const int l15  = lane & 15;
  const int hi   = lane >> 4;

  // corner addresses (clamped, plane-relative) and mask-folded weights
  __shared__ __align__(16) int   s_ad[9 * 64 * 4];   // 9 KB
  __shared__ __align__(16) float s_wt[9 * 64 * 4];   // 9 KB
  __shared__ __align__(16) bf16_t s_B[2][64 * 32];   // ping-pong, 2 x 4 KB

  // ---- precompute bilinear geometry for this image row (cin-invariant) ----
  for (int e = tid; e < 9 * 64; e += 256) {
    int k = e >> 6, n = e & 63;
    size_t pix = (size_t)y * 64 + n;
    float oy = offp[((size_t)(b * 18 + 2 * k)) * HW + pix];
    float ox = offp[((size_t)(b * 18 + 2 * k + 1)) * HW + pix];
    float mk = mskp[((size_t)(b * 9 + k)) * HW + pix];
    int kh = k / 3;
    float py = (float)(y - 1 + kh) + oy;
    float px = (float)(n - 1 + (k - kh * 3)) + ox;
    float fy = floorf(py), fx = floorf(px);
    int y0 = (int)fy, x0 = (int)fx;
    float wy1 = py - fy, wx1 = px - fx;
    float wy0 = 1.0f - wy1, wx0 = 1.0f - wx1;
#pragma unroll
    for (int dy = 0; dy < 2; ++dy)
#pragma unroll
      for (int dx = 0; dx < 2; ++dx) {
        int yy = y0 + dy, xx = x0 + dx;
        bool valid = (yy >= 0) & (yy < 64) & (xx >= 0) & (xx < 64);
        int yi = min(max(yy, 0), 63), xi = min(max(xx, 0), 63);
        float wgt = (dy ? wy1 : wy0) * (dx ? wx1 : wx0);
        int c4 = e * 4 + dy * 2 + dx;
        s_ad[c4] = yi * 64 + xi;
        s_wt[c4] = valid ? (wgt * mk) : 0.0f;
      }
  }

  // ---- wave-uniform (cin,k) bookkeeping, forced into SGPRs ----
  // Within a wave, tid>>6 is constant (= wave>>1), so t = i*4 + tbase and the
  // derived (cin,k) are wave-uniform: keep them scalar via readfirstlane.
  const int n_px    = tid & 63;                                // per-lane pixel
  const int tbase_s = __builtin_amdgcn_readfirstlane(tid >> 6); // scalar
  int cin_i[8], k_i[8];
#pragma unroll
  for (int i = 0; i < 8; ++i) {
    int t = i * 4 + tbase_s;     // < 32, scalar
    int c = (t * 7282) >> 16;    // t/9 for t<32
    cin_i[i] = c;
    k_i[i]   = t - c * 9;
  }

  v8f acc[4] = {};
  const float* xb = xin + (size_t)b * Cin * HW;
  const bf16_t* rp = wbf + (size_t)(wave * 16 + l15) * 32 + hi * 8;

  __syncthreads();

  for (int kc = 0; kc < nchunk; ++kc) {
    bf16_t* sb = s_B[kc & 1];
    // ---- gather 32 K-slices x 64 pixels using precomputed geometry ----
#pragma unroll
    for (int i = 0; i < 8; ++i) {
      int k   = k_i[i];          // scalar
      int cin = cin_i[i];        // scalar
      int e4  = k * 256 + n_px * 4;          // scalar*imm + per-lane
      int4   ad = *(const int4*)(s_ad + e4);
      float4 wt = *(const float4*)(s_wt + e4);
      const float* img = xb + (size_t)cin * HW;   // scalar base
      float v = wt.x * img[ad.x] + wt.y * img[ad.y] +
                wt.z * img[ad.z] + wt.w * img[ad.w];
      sb[n_px * 32 + (i * 4 + tbase_s)] = (bf16_t)v;
      // advance flat index by 32: +3 channels, +5 taps (mod 9), scalar math
      k += 5; cin += 3;
      if (k >= 9) { k -= 9; ++cin; }
      k_i[i] = k; cin_i[i] = cin;
    }
    __syncthreads();   // single barrier: ping-pong makes the 2nd one redundant

    if (kc + 1 < nchunk)
      __builtin_prefetch(rp + (size_t)(kc + 1) * (COUT * 32), 0, 0);

    // ---- A fragment (16x32 bf16) straight from global, chunk-major ----
    // lanes 0-15: row M=l15, K {0..7,16..23}; lanes 16-31: K {8..15,24..31}
    FragBF a;
    const bf16_t* rpc = rp + (size_t)kc * (COUT * 32);
    a.q[0] = *(const uint4*)(rpc);
    a.q[1] = *(const uint4*)(rpc + 16);

    // ---- 4 N-subtiles of 16 pixels ----
#pragma unroll
    for (int nt = 0; nt < 4; ++nt) {
      FragBF bfr;   // lanes 0-15: col N=l15, K {0..7,16..23} ; lanes 16-31: K {8..15,24..31}
      const bf16_t* bp = sb + (size_t)(nt * 16 + l15) * 32 + hi * 8;
      bfr.q[0] = *(const uint4*)(bp);
      bfr.q[1] = *(const uint4*)(bp + 16);
      acc[nt] = __builtin_amdgcn_wmma_f32_16x16x32_bf16(
          false, a.v, false, bfr.v, (short)0, acc[nt], false, false);
    }
  }

  // ---- epilogue: batchnorm (+relu), C layout VGPR r: M = r + 8*hi ----
  float sc[8], sh[8];
#pragma unroll
  for (int r = 0; r < 8; ++r) {
    int ch = wave * 16 + r + 8 * hi;
    float inv = rsqrtf(bnv[ch] + 1e-5f) * bng[ch];
    sc[r] = inv;
    sh[r] = bnb[ch] - bnm[ch] * inv;
  }
#pragma unroll
  for (int nt = 0; nt < 4; ++nt) {
    int col = nt * 16 + l15;
    size_t pix = (size_t)y * 64 + col;
#pragma unroll
    for (int r = 0; r < 8; ++r) {
      int ch = wave * 16 + r + 8 * hi;
      float v = acc[nt][r] * sc[r] + sh[r];
      if (relu) v = fmaxf(v, 0.0f);
      out[((size_t)(b * COUT + ch)) * HW + pix] = v;
    }
  }
}

// ---------------------------------------------------------------------------
// Global mean pool over H*W
// ---------------------------------------------------------------------------
__launch_bounds__(256)
__global__ void pool_kernel(const float* __restrict__ in, float* __restrict__ pooled) {
  int bc = blockIdx.x;                       // b*128 + c
  const float* p = in + (size_t)bc * HW;
  float s = 0.0f;
  for (int i = threadIdx.x; i < HW; i += 256) s += p[i];
  __shared__ float red[256];
  red[threadIdx.x] = s;
  __syncthreads();
  for (int st = 128; st > 0; st >>= 1) {
    if (threadIdx.x < st) red[threadIdx.x] += red[threadIdx.x + st];
    __syncthreads();
  }
  if (threadIdx.x == 0) pooled[bc] = red[0] * (1.0f / (float)HW);
}

// ---------------------------------------------------------------------------
// FC: [16 x 128] x [1000 x 128]^T + bias
// ---------------------------------------------------------------------------
__launch_bounds__(256)
__global__ void fc_kernel(const float* __restrict__ pooled,
                          const float* __restrict__ fw,
                          const float* __restrict__ fb,
                          float* __restrict__ out) {
  int gid = blockIdx.x * 256 + threadIdx.x;
  if (gid >= B_ * 1000) return;
  int n = gid % 1000;
  int b = gid / 1000;
  const float* pv = pooled + b * COUT;
  const float* wr = fw + (size_t)n * COUT;
  float s = fb[n];
#pragma unroll 4
  for (int c = 0; c < COUT; ++c) s += pv[c] * wr[c];
  out[gid] = s;
}

// ---------------------------------------------------------------------------
extern "C" void kernel_launch(void* const* d_in, const int* in_sizes, int n_in,
                              void* d_out, int out_size, void* d_ws, size_t ws_size,
                              hipStream_t stream) {
  const float* x       = (const float*)d_in[0];
  const float* conv1_w = (const float*)d_in[1];
  const float* off1_w  = (const float*)d_in[2];
  const float* mod1_w  = (const float*)d_in[3];
  const float* bn1_g   = (const float*)d_in[4];
  const float* bn1_b   = (const float*)d_in[5];
  const float* bn1_m   = (const float*)d_in[6];
  const float* bn1_v   = (const float*)d_in[7];
  const float* conv2_w = (const float*)d_in[8];
  const float* off2_w  = (const float*)d_in[9];
  const float* mod2_w  = (const float*)d_in[10];
  const float* bn2_g   = (const float*)d_in[11];
  const float* bn2_b   = (const float*)d_in[12];
  const float* bn2_m   = (const float*)d_in[13];
  const float* bn2_v   = (const float*)d_in[14];
  const float* fc_w    = (const float*)d_in[15];
  const float* fc_b    = (const float*)d_in[16];
  float* out = (float*)d_out;

  uintptr_t p = (uintptr_t)d_ws;
  auto carve = [&](size_t bytes) -> void* {
    void* r = (void*)p;
    p += (bytes + 255) & ~(size_t)255;
    return r;
  };
  float*  off_buf = (float*) carve((size_t)B_ * 18 * HW * 4);
  float*  msk_buf = (float*) carve((size_t)B_ * 9 * HW * 4);
  float*  out1    = (float*) carve((size_t)B_ * COUT * HW * 4);
  float*  out2    = (float*) carve((size_t)B_ * COUT * HW * 4);
  bf16_t* w1b     = (bf16_t*)carve((size_t)18 * COUT * 32 * 2);
  bf16_t* w2b     = (bf16_t*)carve((size_t)36 * COUT * 32 * 2);
  float*  pooled  = (float*) carve((size_t)B_ * COUT * 4);

  convert_w<<<(COUT * 576 + 255) / 256, 256, 0, stream>>>(conv1_w, w1b, COUT, 576);
  convert_w<<<(COUT * 1152 + 255) / 256, 256, 0, stream>>>(conv2_w, w2b, COUT, 1152);

  int omblocks = (B_ * 27 * HW) / 256;   // exact

  // layer 1
  offmod_conv<<<omblocks, 256, 0, stream>>>(x, off1_w, mod1_w, off_buf, msk_buf, 64);
  deform_wmma<<<B_ * 64, 256, 0, stream>>>(x, off_buf, msk_buf, w1b,
                                           bn1_g, bn1_b, bn1_m, bn1_v,
                                           out1, 64, 18, 1);
  // layer 2 (reuse off/mask scratch)
  offmod_conv<<<omblocks, 256, 0, stream>>>(out1, off2_w, mod2_w, off_buf, msk_buf, 128);
  deform_wmma<<<B_ * 64, 256, 0, stream>>>(out1, off_buf, msk_buf, w2b,
                                           bn2_g, bn2_b, bn2_m, bn2_v,
                                           out2, 128, 36, 0);
  // head
  pool_kernel<<<B_ * COUT, 256, 0, stream>>>(out2, pooled);
  fc_kernel<<<(B_ * 1000 + 255) / 256, 256, 0, stream>>>(pooled, fc_w, fc_b, out);
}
